// ARMA_25958782337760
// MI455X (gfx1250) — compile-verified, matrix operand order checked by reading, not run on hardware
//
#include <hip/hip_runtime.h>
#include <hip/hip_bf16.h>

typedef unsigned int  u32;
typedef unsigned short u16;
typedef __attribute__((ext_vector_type(16))) __bf16 bfv16;
typedef __attribute__((ext_vector_type(8)))  float  fv8;

// B=64, T=2048, F=256, ARSIZE=8, MASIZE=8
#define NB 64
#define NT 2048
#define NF 256

__device__ __forceinline__ u16 f2bf(float f) {
    u32 u = __float_as_uint(f);
    u32 r = (u + 0x7FFFu + ((u >> 16) & 1u)) >> 16;   // round-to-nearest-even
    return (u16)r;
}

// ---------------- f32 -> bf16 bulk convert ----------------
__global__ void __launch_bounds__(256) cvt_f32_bf16(const float* __restrict__ src,
                                                    u16* __restrict__ dst, int n) {
    for (int i = blockIdx.x * blockDim.x + threadIdx.x; i < n; i += gridDim.x * blockDim.x)
        dst[i] = f2bf(src[i]);
}

// ---------------- pack weight [o=256 rows, K cols] into WMMA-B fragment order --------
// dst layout: tid = ((nt*KT + kt)*32 + lane)*16 + e ; B[k = kt*32 + f(lane,e), o = nt*16 + lane%16]
__global__ void __launch_bounds__(256) pack_w(const float* __restrict__ src,
                                              u16* __restrict__ dst, int K, int KT, int total) {
    for (int tid = blockIdx.x * blockDim.x + threadIdx.x; tid < total;
         tid += gridDim.x * blockDim.x) {
        int e    = tid & 15;
        int lane = (tid >> 4) & 31;
        int frag = tid >> 9;
        int kt   = frag % KT;
        int nt   = frag / KT;
        int o    = nt * 16 + (lane & 15);
        int iin  = (lane < 16) ? ((e < 8) ? e : e + 8)
                               : ((e < 8) ? e + 8 : e + 16);
        int k    = kt * 32 + iin;
        dst[tid] = f2bf(src[(size_t)o * K + k]);
    }
}

// ---------------- MA branch: causal conv as WMMA GEMM, writes tanh(ma) to out --------
// M = B*T (tiles of 16 consecutive t, fixed b), N = 256 (16 tiles), K = 2304 (72 chunks)
__global__ void __launch_bounds__(256) ma_conv_wmma(const u16* __restrict__ xb,
                                                    const u16* __restrict__ Wp,
                                                    const float* __restrict__ bias,
                                                    float* __restrict__ out) {
    const int lane  = threadIdx.x & 31;
    const int gw    = blockIdx.x * 8 + (threadIdx.x >> 5);   // global wave id
    const int mtile = gw >> 2;                               // 8192 m-tiles
    const int strip = gw & 3;                                // 4 n-tiles per strip
    const int b     = mtile >> 7;                            // T/16 = 128 tiles per batch
    const int t0    = (mtile & 127) << 4;
    const int m     = lane & 15;
    const int hi    = (lane >> 4) << 3;                      // 0 or 8

    fv8 acc[4] = {};
    for (int kt = 0; kt < 72; ++kt) {
        const int tap  = kt >> 3;            // 256/32 = 8 chunks per tap
        const int i0   = (kt & 7) << 5;
        const int tsrc = t0 + m - 8 + tap;   // causal left pad
        union { bfv16 v; uint4 q[2]; } a;
        if (tsrc >= 0) {
            const u16* p = xb + ((size_t)b * NT + tsrc) * NF + i0 + hi;
            a.q[0] = *(const uint4*)(p);
            a.q[1] = *(const uint4*)(p + 16);
        } else {
            a.q[0] = make_uint4(0u, 0u, 0u, 0u);
            a.q[1] = make_uint4(0u, 0u, 0u, 0u);
        }
#pragma unroll
        for (int q = 0; q < 4; ++q) {
            const int nt = strip * 4 + q;
            bfv16 bf = *(const bfv16*)(Wp + (((size_t)(nt * 72 + kt) * 32 + lane) << 4));
            acc[q] = __builtin_amdgcn_wmma_f32_16x16x32_bf16(
                false, a.v, false, bf, (short)0, acc[q], false, false);
        }
    }
#pragma unroll
    for (int q = 0; q < 4; ++q) {
        const int nt = strip * 4 + q;
        const int o  = nt * 16 + (lane & 15);
        const float bv = bias[o];
#pragma unroll
        for (int r = 0; r < 8; ++r) {
            const int t = t0 + r + hi;       // C layout: m = r (+8 for hi lanes)
            out[((size_t)b * NT + t) * NF + o] = tanhf(acc[r == r ? q : q][r] + bv);
        }
    }
}

// ---------------- AR branch: persistent single-WGP sequential scan ----------------
// LDS ring: 8 slots x packed-A layout [4 mtiles][8 ktiles][32 lanes][16 e] bf16 = 256 KB
__global__ void __launch_bounds__(1024) ar_scan_wmma(const u16* __restrict__ Wp,
                                                     const float* __restrict__ b_ar,
                                                     float* __restrict__ out) {
    extern __shared__ u16 lds[];             // 262144 bytes
    const int tid  = threadIdx.x;
    const int lane = tid & 31;
    const int wave = tid >> 5;
    const int mt   = wave & 3;               // batch tile (16 rows of b)
    const int nt0  = (wave >> 2) * 2;        // two output-feature tiles per wave
    const int nt1  = nt0 + 1;

    for (int i = tid; i < 65536; i += 1024) ((u32*)lds)[i] = 0u;   // zero history
    __syncthreads();

    const int col = lane & 15;
    const int bhi = (lane >> 4) << 3;        // 0 or 8
    const int o0  = nt0 * 16 + col;
    const int o1  = nt1 * 16 + col;
    const float bias0 = b_ar[o0];
    const float bias1 = b_ar[o1];

    // precompute scatter params for writing y into packed-A ring layout
    const int it_0 = o0 >> 5, iin0 = o0 & 31;
    const int it_1 = o1 >> 5, iin1 = o1 & 31;
    const int lg0 = (iin0 >> 3) & 1, e0 = (iin0 & 7) + ((iin0 >> 4) << 3);
    const int lg1 = (iin1 >> 3) & 1, e1 = (iin1 & 7) + ((iin1 >> 4) << 3);

    for (int t = 0; t < NT; ++t) {
        // prefetch ma(t) for our two tiles (overlaps with the GEMM below)
        float ma0[8], ma1[8];
#pragma unroll
        for (int r = 0; r < 8; ++r) {
            const size_t base = (((size_t)(mt * 16 + r + bhi)) * NT + t) * NF;
            ma0[r] = out[base + o0];
            ma1[r] = out[base + o1];
        }

        fv8 acc0 = {}, acc1 = {};
        for (int s = 0; s < 8; ++s) {
            const int j = (s - t) & 7;       // slot s holds y_{t-8+j}
            const u16* ab  = lds + (((size_t)((s * 4 + mt) * 8) * 32 + lane) << 4);
            const u16* wb0 = Wp + (((size_t)(nt0 * 64 + j * 8) * 32 + lane) << 4);
            const u16* wb1 = Wp + (((size_t)(nt1 * 64 + j * 8) * 32 + lane) << 4);
#pragma unroll
            for (int it = 0; it < 8; ++it) {
                bfv16 a  = *(const bfv16*)(ab  + (it << 9));
                bfv16 b0 = *(const bfv16*)(wb0 + (it << 9));
                bfv16 b1 = *(const bfv16*)(wb1 + (it << 9));
                acc0 = __builtin_amdgcn_wmma_f32_16x16x32_bf16(
                    false, a, false, b0, (short)0, acc0, false, false);
                acc1 = __builtin_amdgcn_wmma_f32_16x16x32_bf16(
                    false, a, false, b1, (short)0, acc1, false, false);
            }
        }
        __syncthreads();                     // everyone done reading slot t&7

        const int s2 = t & 7;
        const int sb = (s2 * 4 + mt) * 8;    // ring tile base for this wave's batch rows
#pragma unroll
        for (int r = 0; r < 8; ++r) {
            const int bb = mt * 16 + r + bhi;
            const size_t base = ((size_t)bb * NT + t) * NF;
            const float y0 = tanhf(ma0[r] + tanhf(acc0[r] + bias0));
            const float y1 = tanhf(ma1[r] + tanhf(acc1[r] + bias1));
            out[base + o0] = y0;
            out[base + o1] = y1;
            const int bl = r + bhi;          // bb & 15
            lds[(((sb + it_0) * 32 + (bl | (lg0 << 4))) << 4) + e0] = f2bf(y0);
            lds[(((sb + it_1) * 32 + (bl | (lg1 << 4))) << 4) + e1] = f2bf(y1);
        }
        __syncthreads();                     // ring update visible before next step
    }
}

extern "C" void kernel_launch(void* const* d_in, const int* in_sizes, int n_in,
                              void* d_out, int out_size, void* d_ws, size_t ws_size,
                              hipStream_t stream) {
    const float* x    = (const float*)d_in[0];   // [64,2048,256]
    const float* W_ar = (const float*)d_in[1];   // [256,2048]
    const float* b_ar = (const float*)d_in[2];   // [256]
    const float* W_ma = (const float*)d_in[3];   // [256,2304]
    const float* b_ma = (const float*)d_in[4];   // [256]
    float* out = (float*)d_out;

    u16* xb    = (u16*)d_ws;                     // 33,554,432 elems
    u16* Wp_ma = xb + 33554432;                  //    589,824 elems (16 nt * 72 kt * 512)
    u16* Wp_ar = Wp_ma + 589824;                 //    524,288 elems (16 nt * 64 kt * 512)

    cvt_f32_bf16<<<8192, 256, 0, stream>>>(x, xb, 33554432);
    pack_w<<<2304, 256, 0, stream>>>(W_ma, Wp_ma, 2304, 72, 589824);
    pack_w<<<2048, 256, 0, stream>>>(W_ar, Wp_ar, 2048, 64, 524288);
    ma_conv_wmma<<<4096, 256, 0, stream>>>(xb, Wp_ma, b_ma, out);
    ar_scan_wmma<<<1, 1024, 262144, stream>>>(Wp_ar, b_ar, out);
}
